// ClusterReasoning_10634339025513
// MI455X (gfx1250) — matrix operand order repaired and checked by reading, not compiled, not run on hardware
//
#include <hip/hip_runtime.h>
#include <hip/hip_bf16.h>
#include <math.h>

// ---------------------------------------------------------------------------
// Visual branch dims (text branch is provably dead: align output == 0)
// ---------------------------------------------------------------------------
#define VB   8
#define VS   900
#define VH   1024
#define VNH  16
#define VD   64
#define VI   3072
#define VM   (VB * VS)          // 7200 rows (divisible by 16: 450 subtiles)
#define NEGBIG  (-1e9f)
#define LN_EPS  (1e-12f)

typedef __attribute__((ext_vector_type(16))) _Float16 v16h;
typedef __attribute__((ext_vector_type(8)))  float    v8f;

#define WMMA_F16(a, b, c) \
  __builtin_amdgcn_wmma_f32_16x16x32_f16(false, (a), false, (b), (short)0, (c), false, false)

// ---------------------------------------------------------------------------
// Fragment-order storage (CDNA5 ISA wave32 layouts).
// A 16x32 subtile / B 32x16 subtile = 512 halves stored as [lane][e0..15],
// so a fragment load is one contiguous 32B v16h read per lane.
//   A element (m,k): lane = (m&15) + ((k&8)?16:0)
//                    e    = ((k&16)?8:0) + (((k>>1)&3)<<1) + (k&1)
//   B element (k,n): lane = (n&15) + ((k&8)?16:0) ; e = ((k&16)?8:0) + (k&7)
// ---------------------------------------------------------------------------
__device__ inline int frag_a_idx(int m, int k) {
  const int lane = (m & 15) + ((k & 8) ? 16 : 0);
  const int e    = ((k & 16) ? 8 : 0) + (((k >> 1) & 3) << 1) + (k & 1);
  return lane * 16 + e;
}
__device__ inline int frag_b_idx(int k, int n) {
  const int lane = (n & 15) + ((k & 8) ? 16 : 0);
  return lane * 16 + ((k & 16) ? 8 : 0) + (k & 7);
}
__device__ inline v16h load_frag(const _Float16* p, int lane) {
  return *(const v16h*)(p + lane * 16);
}

// reductions across the 16-lane half-groups (rows of the C layout)
__device__ inline float rowmax16(float x) {
#pragma unroll
  for (int off = 1; off < 16; off <<= 1) x = fmaxf(x, __shfl_xor(x, off, 32));
  return x;
}
__device__ inline float rowsum16(float x) {
#pragma unroll
  for (int off = 1; off < 16; off <<= 1) x += __shfl_xor(x, off, 32);
  return x;
}

// ---------------------------------------------------------------------------
// pack_a: X f32 [M,K] -> Xf f16 fragment panels [mt][kt][512]
// one 256-thr block per (mt,kt) subtile; coalesced reads, scattered f16 writes
// ---------------------------------------------------------------------------
__global__ __launch_bounds__(256)
void pack_a_kernel(const float* __restrict__ X, _Float16* __restrict__ Xf, int KT) {
  const int mt = blockIdx.x / KT, kt = blockIdx.x % KT;
  const int K = KT << 5;
  _Float16* dst = Xf + (size_t)blockIdx.x * 512;
#pragma unroll
  for (int t = 0; t < 2; ++t) {
    const int li = threadIdx.x + t * 256;     // 0..511
    const int m = li >> 5, k = li & 31;
    dst[frag_a_idx(m, k)] = (_Float16)X[(size_t)(mt * 16 + m) * K + kt * 32 + k];
  }
}

// pack_b: W f32 [K,N] -> Wf f16 fragment panels [kt][nt][512]
__global__ __launch_bounds__(256)
void pack_b_kernel(const float* __restrict__ W, _Float16* __restrict__ Wf, int NT) {
  const int kt = blockIdx.x / NT, nt = blockIdx.x % NT;
  const int N = NT << 4;
  _Float16* dst = Wf + (size_t)blockIdx.x * 512;
#pragma unroll
  for (int t = 0; t < 2; ++t) {
    const int li = threadIdx.x + t * 256;     // 0..511
    const int k = li >> 4, n = li & 15;
    dst[frag_b_idx(k, n)] = (_Float16)W[(size_t)(kt * 32 + k) * N + nt * 16 + n];
  }
}

// ---------------------------------------------------------------------------
// GEMM on fragment panels: Y[M,N] = Xf * Wf + bias (optional exact GELU).
// block = 256 thr (8 waves), tile 128x64; no LDS, no barriers: every
// fragment is a coalesced global b128-pair load, reuse comes from L2.
// ---------------------------------------------------------------------------
#define BM 128
#define BN 64

__global__ __launch_bounds__(256)
void gemm_frag_kernel(const _Float16* __restrict__ Xf, const _Float16* __restrict__ Wf,
                      const float* __restrict__ bias, float* __restrict__ Y,
                      int M, int N, int K, int gelu) {
  const int KT = K >> 5, NT = N >> 4, MT = M >> 4;
  const int tid = threadIdx.x, lane = tid & 31, w = tid >> 5;
  const int wm = (w >> 1) * 32;               // 0,32,64,96
  const int wn = (w & 1) * 32;                // 0,32
  const int m0 = blockIdx.y * BM;
  const int n0 = blockIdx.x * BN;

  int mt0 = (m0 + wm) >> 4;                   // clamp: garbage rows are never stored
  int mt1 = mt0 + 1;
  mt0 = mt0 < MT - 1 ? mt0 : MT - 1;
  mt1 = mt1 < MT - 1 ? mt1 : MT - 1;
  const int nt0 = (n0 + wn) >> 4, nt1 = nt0 + 1;

  const _Float16* ax0 = Xf + (size_t)mt0 * KT * 512 + lane * 16;
  const _Float16* ax1 = Xf + (size_t)mt1 * KT * 512 + lane * 16;
  const _Float16* bx0 = Wf + (size_t)nt0 * 512 + lane * 16;
  const _Float16* bx1 = Wf + (size_t)nt1 * 512 + lane * 16;

  v8f acc[2][2];
#pragma unroll
  for (int a = 0; a < 2; ++a)
#pragma unroll
    for (int b = 0; b < 2; ++b)
#pragma unroll
      for (int e = 0; e < 8; ++e) acc[a][b][e] = 0.f;

#pragma unroll 2
  for (int kt = 0; kt < KT; ++kt) {
    const v16h a0 = *(const v16h*)(ax0 + (size_t)kt * 512);
    const v16h a1 = *(const v16h*)(ax1 + (size_t)kt * 512);
    const v16h b0 = *(const v16h*)(bx0 + (size_t)kt * NT * 512);
    const v16h b1 = *(const v16h*)(bx1 + (size_t)kt * NT * 512);
    acc[0][0] = WMMA_F16(a0, b0, acc[0][0]);
    acc[0][1] = WMMA_F16(a0, b1, acc[0][1]);
    acc[1][0] = WMMA_F16(a1, b0, acc[1][0]);
    acc[1][1] = WMMA_F16(a1, b1, acc[1][1]);
  }

  const int cn = lane & 15;
  const int mh = (lane & 16) ? 8 : 0;
#pragma unroll
  for (int bt = 0; bt < 2; ++bt) {
    const int n = n0 + wn + bt * 16 + cn;
    const float bv = bias ? bias[n] : 0.f;
#pragma unroll
    for (int at = 0; at < 2; ++at) {
#pragma unroll
      for (int i = 0; i < 8; ++i) {
        const int m = m0 + wm + at * 16 + mh + i;
        if (m < M) {
          float v = acc[at][bt][i] + bv;
          if (gelu) v = 0.5f * v * (1.f + erff(v * 0.70710678118f));
          Y[(size_t)m * N + n] = v;
        }
      }
    }
  }
}

// ---------------------------------------------------------------------------
// Flash attention (visual): per (b,head), 16-row Q tiles, 32-key steps.
// bias(j) = (cluster[b,j]==pass) ? 0 : -1e9 ; scale = 1/sqrt(64).
// block = 128 thr (4 waves, one Q tile each); K/V staged in LDS in
// fragment order -> all ds reads are contiguous b128 pairs.
// ---------------------------------------------------------------------------
__global__ __launch_bounds__(128)
void flash_attn_kernel(const float* __restrict__ Q, const float* __restrict__ Kk,
                       const float* __restrict__ Vv, const int* __restrict__ cluster,
                       int pass, float* __restrict__ O) {
  __shared__ _Float16 Qs[4][2][512];    // per-wave A frags, chunk = dd>>5
  __shared__ _Float16 Ks[2][2][512];    // B frags of K^T: [dd-chunk][j-tile]
  __shared__ _Float16 Vs[4][512];       // B frags of V: [dd-tile]
  __shared__ _Float16 Ps[4][512];       // per-wave P as A frag

  const int tid = threadIdx.x, lane = tid & 31, w = tid >> 5;
  const int b = blockIdx.y / VNH, head = blockIdx.y % VNH;
  const int hoff = head * VD;
  const int qt = blockIdx.x * 4 + w;

#pragma unroll
  for (int t = 0; t < 32; ++t) {        // stage all 4 Q tiles (frag order)
    const int idx = tid + t * 128;      // 0..4095
    const int wv = idx >> 10, rem = idx & 1023;
    const int m = rem >> 6, dd = rem & 63;
    const int qrow = (blockIdx.x * 4 + wv) * 16 + m;
    const float v = (qrow < VS) ? Q[(size_t)(b * VS + qrow) * VH + hoff + dd] : 0.f;
    Qs[wv][dd >> 5][frag_a_idx(m, dd & 31)] = (_Float16)v;
  }
  __syncthreads();

  const v16h aq0 = load_frag(Qs[w][0], lane);
  const v16h aq1 = load_frag(Qs[w][1], lane);

  v8f oacc[4];
#pragma unroll
  for (int t = 0; t < 4; ++t)
#pragma unroll
    for (int e = 0; e < 8; ++e) oacc[t][e] = 0.f;
  float mrun[8], lrun[8];
#pragma unroll
  for (int i = 0; i < 8; ++i) { mrun[i] = -1e30f; lrun[i] = 0.f; }

  const float scale = 0.125f;           // 1/sqrt(64)
  const int cn = lane & 15;
  const int mh = (lane & 16) ? 8 : 0;

  for (int j0 = 0; j0 < VS; j0 += 32) {
#pragma unroll
    for (int t = 0; t < 16; ++t) {      // stage K^T and V tiles (frag order)
      const int idx = tid + t * 128;    // 0..2047
      const int dd = idx >> 5, jj = idx & 31;
      const int jg = j0 + jj;
      Ks[dd >> 5][jj >> 4][frag_b_idx(dd & 31, jj & 15)] =
          (_Float16)((jg < VS) ? Kk[(size_t)(b * VS + jg) * VH + hoff + dd] : 0.f);
      const int jj2 = idx >> 6, dd2 = idx & 63;
      const int jg2 = j0 + jj2;
      Vs[dd2 >> 4][frag_b_idx(jj2, dd2 & 15)] =
          (_Float16)((jg2 < VS) ? Vv[(size_t)(b * VS + jg2) * VH + hoff + dd2] : 0.f);
    }
    __syncthreads();

    v8f s[2];
#pragma unroll
    for (int jt = 0; jt < 2; ++jt) {    // two 16x16 score tiles
#pragma unroll
      for (int e = 0; e < 8; ++e) s[jt][e] = 0.f;
      const v16h bk0 = load_frag(Ks[0][jt], lane);
      const v16h bk1 = load_frag(Ks[1][jt], lane);
      s[jt] = WMMA_F16(aq0, bk0, s[jt]);
      s[jt] = WMMA_F16(aq1, bk1, s[jt]);
    }
#pragma unroll
    for (int jt = 0; jt < 2; ++jt) {    // scale + cluster mask bias
      const int jg = j0 + jt * 16 + cn;
      float bias = NEGBIG;
      if (jg < VS && cluster[b * VS + jg] == pass) bias = 0.f;
#pragma unroll
      for (int i = 0; i < 8; ++i) s[jt][i] = s[jt][i] * scale + bias;
    }

    float fac[8];
#pragma unroll
    for (int i = 0; i < 8; ++i) {       // online softmax per row
      float t = fmaxf(s[0][i], s[1][i]);
      t = rowmax16(t);
      const float mnew = fmaxf(mrun[i], t);
      fac[i] = __expf(mrun[i] - mnew);
      s[0][i] = __expf(s[0][i] - mnew);
      s[1][i] = __expf(s[1][i] - mnew);
      const float rs = rowsum16(s[0][i] + s[1][i]);
      lrun[i] = lrun[i] * fac[i] + rs;
      mrun[i] = mnew;
    }
#pragma unroll
    for (int t = 0; t < 4; ++t)
#pragma unroll
      for (int i = 0; i < 8; ++i) oacc[t][i] *= fac[i];

#pragma unroll
    for (int i = 0; i < 8; ++i) {       // P: C layout -> A frag layout in LDS
      Ps[w][frag_a_idx(mh + i, cn)]      = (_Float16)s[0][i];
      Ps[w][frag_a_idx(mh + i, 16 + cn)] = (_Float16)s[1][i];
    }
    const v16h ap = load_frag(Ps[w], lane);
#pragma unroll
    for (int t = 0; t < 4; ++t) {       // O += P(16x32) x V(32x64)
      const v16h bv = load_frag(Vs[t], lane);
      oacc[t] = WMMA_F16(ap, bv, oacc[t]);
    }
    __syncthreads();
  }

  if (qt < (VS + 15) / 16) {
#pragma unroll
    for (int i = 0; i < 8; ++i) {
      const int srow = qt * 16 + mh + i;
      if (srow < VS) {
        const float inv = 1.f / lrun[i];
#pragma unroll
        for (int t = 0; t < 4; ++t)
          O[(size_t)(b * VS + srow) * VH + hoff + t * 16 + cn] = oacc[t][i] * inv;
      }
    }
  }
}

// ---------------------------------------------------------------------------
// LayerNorm kernels (H=1024, one 256-thread block per row)
// ---------------------------------------------------------------------------
__device__ inline float blk_reduce_sum(float v, float* sh) {
#pragma unroll
  for (int off = 16; off > 0; off >>= 1) v += __shfl_xor(v, off, 32);
  const int w = threadIdx.x >> 5, nw = blockDim.x >> 5;
  if ((threadIdx.x & 31) == 0) sh[w] = v;
  __syncthreads();
  float t = (threadIdx.x < nw) ? sh[threadIdx.x] : 0.f;
#pragma unroll
  for (int off = 16; off > 0; off >>= 1) t += __shfl_xor(t, off, 32);
  if (threadIdx.x == 0) sh[0] = t;
  __syncthreads();
  const float r = sh[0];
  __syncthreads();
  return r;
}

// e = LN(vs + pos_emb[2+s] + type)
__global__ __launch_bounds__(256)
void embed_ln_kernel(const float* __restrict__ X, const float* __restrict__ pos,
                     const float* __restrict__ typev, const float* __restrict__ g,
                     const float* __restrict__ beta, float* __restrict__ out) {
  __shared__ float sh[8];
  const int row = blockIdx.x;
  const int s = row % VS;
  float x[4];
  float sum = 0.f;
#pragma unroll
  for (int t = 0; t < 4; ++t) {
    const int c = threadIdx.x + t * 256;
    x[t] = X[(size_t)row * VH + c] + pos[(size_t)(s + 2) * VH + c] + typev[c];
    sum += x[t];
  }
  const float mean = blk_reduce_sum(sum, sh) * (1.f / VH);
  float v2 = 0.f;
#pragma unroll
  for (int t = 0; t < 4; ++t) { const float d = x[t] - mean; v2 += d * d; }
  const float rstd = rsqrtf(blk_reduce_sum(v2, sh) * (1.f / VH) + LN_EPS);
#pragma unroll
  for (int t = 0; t < 4; ++t) {
    const int c = threadIdx.x + t * 256;
    out[(size_t)row * VH + c] = (x[t] - mean) * rstd * g[c] + beta[c];
  }
}

// out = LN(a + b)
__global__ __launch_bounds__(256)
void res_ln_kernel(const float* __restrict__ A, const float* __restrict__ Bv,
                   const float* __restrict__ g, const float* __restrict__ beta,
                   float* __restrict__ out) {
  __shared__ float sh[8];
  const int row = blockIdx.x;
  float x[4];
  float sum = 0.f;
#pragma unroll
  for (int t = 0; t < 4; ++t) {
    const int c = threadIdx.x + t * 256;
    x[t] = A[(size_t)row * VH + c] + Bv[(size_t)row * VH + c];
    sum += x[t];
  }
  const float mean = blk_reduce_sum(sum, sh) * (1.f / VH);
  float v2 = 0.f;
#pragma unroll
  for (int t = 0; t < 4; ++t) { const float d = x[t] - mean; v2 += d * d; }
  const float rstd = rsqrtf(blk_reduce_sum(v2, sh) * (1.f / VH) + LN_EPS);
#pragma unroll
  for (int t = 0; t < 4; ++t) {
    const int c = threadIdx.x + t * 256;
    out[(size_t)row * VH + c] = (x[t] - mean) * rstd * g[c] + beta[c];
  }
}

__global__ void zero_kernel(float* __restrict__ p, size_t n) {
  size_t i = (size_t)blockIdx.x * blockDim.x + threadIdx.x;
  const size_t stride = (size_t)gridDim.x * blockDim.x;
  for (; i < n; i += stride) p[i] = 0.f;
}

// ---------------------------------------------------------------------------
// Host orchestration.
// Input flattening (setup_inputs insertion order, nested dicts recursive):
//   0 vision_state, 1 alignment_state, 2 textual_ids, 3 cluster_state,
//   4..28 text_params (25 leaves, all dead code),
//   29.. visual_params: pos_emb, type_emb, emb_ln_g, emb_ln_b,
//        Wq,bq,Wk,bk,Wv,bv,Wo,bo,ln1_g,ln1_b,Wi,bi,Wo2,bo2,ln2_g,ln2_b
// ---------------------------------------------------------------------------
extern "C" void kernel_launch(void* const* d_in, const int* in_sizes, int n_in,
                              void* d_out, int out_size, void* d_ws, size_t ws_size,
                              hipStream_t stream) {
  (void)in_sizes; (void)n_in; (void)out_size; (void)ws_size;

  const float* vision  = (const float*)d_in[0];
  const int*   cluster = (const int*)d_in[3];
  const int VPB = 29;
  const float* pos   = (const float*)d_in[VPB + 0];
  const float* typev = (const float*)d_in[VPB + 1];
  const float* elng  = (const float*)d_in[VPB + 2];
  const float* elnb  = (const float*)d_in[VPB + 3];
  const float* Wq  = (const float*)d_in[VPB + 4];
  const float* bq  = (const float*)d_in[VPB + 5];
  const float* Wk  = (const float*)d_in[VPB + 6];
  const float* bk  = (const float*)d_in[VPB + 7];
  const float* Wv  = (const float*)d_in[VPB + 8];
  const float* bv  = (const float*)d_in[VPB + 9];
  const float* Wo  = (const float*)d_in[VPB + 10];
  const float* bo  = (const float*)d_in[VPB + 11];
  const float* l1g = (const float*)d_in[VPB + 12];
  const float* l1b = (const float*)d_in[VPB + 13];
  const float* Wi  = (const float*)d_in[VPB + 14];
  const float* bi  = (const float*)d_in[VPB + 15];
  const float* Wo2 = (const float*)d_in[VPB + 16];
  const float* bo2 = (const float*)d_in[VPB + 17];
  const float* l2g = (const float*)d_in[VPB + 18];
  const float* l2b = (const float*)d_in[VPB + 19];

  float* ws = (float*)d_ws;
  const size_t T = (size_t)VM * VH;       // 7,372,800 floats
  float* Abuf = ws;                       // vs working buffer
  float* C1   = ws + 1 * T;               // q -> h
  float* C2   = ws + 2 * T;               // k
  float* C3   = ws + 3 * T;               // v -> o
  float* E    = ws + 4 * T;               // e (LN'd embed)
  float* Dd   = ws + 5 * T;               // ctx -> ffn-out
  float* Mid  = C2;                       // FFN mid f32 aliases C2..E (3T = VM*VI)

  // f16 fragment-panel region
  _Float16* hp  = (_Float16*)(ws + 6 * T);
  _Float16* Pa  = hp;                     // packed activations, VM*VI halves max
  _Float16* Wqf = Pa  + (size_t)VM * VI;
  _Float16* Wkf = Wqf + (size_t)VH * VH;
  _Float16* Wvf = Wkf + (size_t)VH * VH;
  _Float16* Wof = Wvf + (size_t)VH * VH;
  _Float16* Wif = Wof + (size_t)VH * VH;
  _Float16* Wo2f = Wif + (size_t)VH * VI;

  float* out_vs    = (float*)d_out;
  float* out_align = out_vs + T;

  // align output is provably zero (see analysis): write zeros
  zero_kernel<<<2048, 256, 0, stream>>>(out_align, (size_t)VB * VS * VS);

  // pack all weights to f16 fragment panels (reused across 5 passes)
  pack_b_kernel<<<(VH / 32) * (VH / 16), 256, 0, stream>>>(Wq,  Wqf,  VH / 16);
  pack_b_kernel<<<(VH / 32) * (VH / 16), 256, 0, stream>>>(Wk,  Wkf,  VH / 16);
  pack_b_kernel<<<(VH / 32) * (VH / 16), 256, 0, stream>>>(Wv,  Wvf,  VH / 16);
  pack_b_kernel<<<(VH / 32) * (VH / 16), 256, 0, stream>>>(Wo,  Wof,  VH / 16);
  pack_b_kernel<<<(VH / 32) * (VI / 16), 256, 0, stream>>>(Wi,  Wif,  VI / 16);
  pack_b_kernel<<<(VI / 32) * (VH / 16), 256, 0, stream>>>(Wo2, Wo2f, VH / 16);

  const int MT = VM / 16;                 // 450
  const dim3 gH(VH / BN, (VM + BM - 1) / BM);
  const dim3 gI(VI / BN, (VM + BM - 1) / BM);
  const dim3 gF(((VS + 15) / 16 + 3) / 4, VB * VNH);

  for (int pass = 0; pass < 5; ++pass) {
    const float* src = pass ? Abuf : vision;
    embed_ln_kernel<<<VM, 256, 0, stream>>>(src, pos, typev, elng, elnb, E);

    pack_a_kernel<<<MT * (VH / 32), 256, 0, stream>>>(E, Pa, VH / 32);
    gemm_frag_kernel<<<gH, 256, 0, stream>>>(Pa, Wqf, bq, C1, VM, VH, VH, 0);
    gemm_frag_kernel<<<gH, 256, 0, stream>>>(Pa, Wkf, bk, C2, VM, VH, VH, 0);
    gemm_frag_kernel<<<gH, 256, 0, stream>>>(Pa, Wvf, bv, C3, VM, VH, VH, 0);

    flash_attn_kernel<<<gF, 128, 0, stream>>>(C1, C2, C3, cluster, pass, Dd);

    pack_a_kernel<<<MT * (VH / 32), 256, 0, stream>>>(Dd, Pa, VH / 32);
    gemm_frag_kernel<<<gH, 256, 0, stream>>>(Pa, Wof, bo, C3, VM, VH, VH, 0);
    res_ln_kernel<<<VM, 256, 0, stream>>>(C3, E, l1g, l1b, C1);            // h

    pack_a_kernel<<<MT * (VH / 32), 256, 0, stream>>>(C1, Pa, VH / 32);
    gemm_frag_kernel<<<gI, 256, 0, stream>>>(Pa, Wif, bi, Mid, VM, VI, VH, 1);

    pack_a_kernel<<<MT * (VI / 32), 256, 0, stream>>>(Mid, Pa, VI / 32);
    gemm_frag_kernel<<<gH, 256, 0, stream>>>(Pa, Wo2f, bo2, Dd, VM, VH, VI, 0);

    res_ln_kernel<<<VM, 256, 0, stream>>>(Dd, C1, l2g, l2b,
                                          (pass == 4) ? out_vs : Abuf);
  }
}
